// SelfCorrelatingRecursiveAttn_57655640981935
// MI455X (gfx1250) — matrix operand order repaired and checked by reading, not compiled, not run on hardware
//
#include <hip/hip_runtime.h>
#include <hip/hip_bf16.h>

typedef __bf16 bf16;
typedef __attribute__((ext_vector_type(16))) __bf16 v16bf;
typedef __attribute__((ext_vector_type(8)))  __bf16 v8bf;
typedef __attribute__((ext_vector_type(8)))  float  v8f;

#define DEV __device__ __forceinline__

static constexpr int Bk = 2, Nk = 2048, Ck = 768, Hk = 12, Dk = 64;
static constexpr int Mk = Bk * Nk;     // 4096 gemm rows (m = b*N + n)
static constexpr int C3 = 3 * Ck;      // 2304
static constexpr int BH = Bk * Hk;     // 24

// ---------------- WMMA fragment helpers (wave32, 16x16x32 bf16) -------------
// A (16xK=32): lane L holds row M=L%16; elems e<8 -> K = hi*8 + e,
//              elems e>=8 -> K = 16 + hi*8 + (e-8)   (hi = L/16)
DEV v16bf load_a_frag(const bf16* p /* row base + k0 + hi*8 */) {
  union { v16bf v; v8bf h[2]; } f;
  f.h[0] = *(const v8bf*)(p);
  f.h[1] = *(const v8bf*)(p + 16);
  return f.v;
}
// B (K=32 x 16): lane L holds col N=L%16; elems e<8 -> K = hi*8 + e,
//              elems e>=8 -> K = 16 + hi*8 + (e-8)
DEV v16bf load_b_frag(const bf16* p /* col base + k0 + hi*8 */) {
  union { v16bf v; v8bf h[2]; } f;
  f.h[0] = *(const v8bf*)(p);
  f.h[1] = *(const v8bf*)(p + 16);
  return f.v;
}
DEV v8f wmma_bf16(v16bf a, v16bf b, v8f c) {
  return __builtin_amdgcn_wmma_f32_16x16x32_bf16(false, a, false, b,
                                                 (short)0, c, false, false);
}

// ---------------- elementwise f32 -> bf16 -----------------------------------
__global__ void f2bf_kernel(const float* __restrict__ in,
                            bf16* __restrict__ out, int n) {
  int i = blockIdx.x * 256 + threadIdx.x;
  if (i < n) out[i] = (bf16)in[i];
}

// v [BH, N, D] -> vT [BH, D, N]  (so P@V B-operand reads contiguous keys)
__global__ void vtrans_kernel(const bf16* __restrict__ v,
                              bf16* __restrict__ vT) {
  int i = blockIdx.x * 256 + threadIdx.x;      // over BH*N*D
  int d = i & 63;
  int n = (i >> 6) & (Nk - 1);
  int bh = i >> 17;
  vT[((size_t)bh * Dk + d) * Nk + n] = v[i];
}

// row-wise L2 normalize p1 [BH*N, 64] f32 -> bf16 (one wave per row)
__global__ __launch_bounds__(256)
void normalize_kernel(const float* __restrict__ p1, bf16* __restrict__ tb) {
  int row  = blockIdx.x * 8 + (threadIdx.x >> 5);
  int lane = threadIdx.x & 31;
  const float* r = p1 + (size_t)row * Dk;
  float a = r[lane], b2 = r[lane + 32];
  float ss = a * a + b2 * b2;
  ss += __shfl_xor(ss, 1);  ss += __shfl_xor(ss, 2);
  ss += __shfl_xor(ss, 4);  ss += __shfl_xor(ss, 8);
  ss += __shfl_xor(ss, 16);
  float inv = 1.0f / fmaxf(sqrtf(ss), 1e-12f);
  bf16* o = tb + (size_t)row * Dk;
  o[lane]      = (bf16)(a  * inv);
  o[lane + 32] = (bf16)(b2 * inv);
}

// ---------------- tiled GEMM: out = A[M,K] @ W[N,K]^T + bias ----------------
// 8 waves/block, each wave owns a 32x32 output tile (2x2 WMMA tiles),
// software-pipelined fragment loads (next K-block loads issued before the
// current WMMAs so loadcnt waits overlap XDL work).
// MODE 0: QKV, scatter bf16 into q/k/v [BH,N,D]
// MODE 1: proj1, scatter f32 into p1 [BH,N,D]
// MODE 2: proj2, write f32 [B,N,C] row-major (row m = b*N+n)
template<int MODE>
__global__ __launch_bounds__(256)
void gemm_kernel(const bf16* __restrict__ A, const bf16* __restrict__ W,
                 const float* __restrict__ bias, int K,
                 bf16* __restrict__ qo, bf16* __restrict__ ko,
                 bf16* __restrict__ vo, float* __restrict__ fo) {
  const int lane    = threadIdx.x & 31;
  const int wv      = threadIdx.x >> 5;          // 8 waves / block
  const int laneRow = lane & 15;
  const int hi      = lane >> 4;
  const int row0 = blockIdx.y * 64  + (wv & 1) * 32;   // block: 64 x 128
  const int col0 = blockIdx.x * 128 + (wv >> 1) * 32;

  const bf16* ar0 = A + (size_t)(row0 + laneRow) * K + hi * 8;
  const bf16* ar1 = ar0 + (size_t)16 * K;
  const bf16* wr0 = W + (size_t)(col0 + laneRow) * K + hi * 8;
  const bf16* wr1 = wr0 + (size_t)16 * K;

  v8f c00 = {}, c01 = {}, c10 = {}, c11 = {};
  v16bf a0 = load_a_frag(ar0), a1 = load_a_frag(ar1);
  v16bf b0 = load_b_frag(wr0), b1 = load_b_frag(wr1);

  for (int k0 = 32; k0 <= K; k0 += 32) {
    // issue next K-block loads first (wrap to 0 on the last trip: harmless,
    // keeps the pipeline branch-free)
    const int kn = (k0 < K) ? k0 : 0;
    v16bf na0 = load_a_frag(ar0 + kn);
    v16bf na1 = load_a_frag(ar1 + kn);
    v16bf nb0 = load_b_frag(wr0 + kn);
    v16bf nb1 = load_b_frag(wr1 + kn);
    if (k0 + 32 < K) {
      __builtin_prefetch(ar0 + k0 + 32, 0, 3);
      __builtin_prefetch(ar1 + k0 + 32, 0, 3);
      __builtin_prefetch(wr0 + k0 + 32, 0, 3);
      __builtin_prefetch(wr1 + k0 + 32, 0, 3);
    }
    c00 = wmma_bf16(a0, b0, c00);
    c01 = wmma_bf16(a0, b1, c01);
    c10 = wmma_bf16(a1, b0, c10);
    c11 = wmma_bf16(a1, b1, c11);
    a0 = na0; a1 = na1; b0 = nb0; b1 = nb1;
  }

  // epilogue over the 2x2 tile grid (D-layout: lane=col, vgpr r -> row)
#pragma unroll
  for (int ti = 0; ti < 2; ++ti) {
#pragma unroll
    for (int tj = 0; tj < 2; ++tj) {
      const v8f& acc = (ti == 0) ? (tj == 0 ? c00 : c01)
                                 : (tj == 0 ? c10 : c11);
#pragma unroll
      for (int r = 0; r < 8; ++r) {
        int gm = row0 + ti * 16 + r + 8 * hi;
        int gc = col0 + tj * 16 + laneRow;
        float val = acc[r] + bias[gc];
        if (MODE == 0) {
          int part = gc / Ck;                    // 0=q 1=k 2=v
          int cc = gc - part * Ck;
          int h = cc >> 6, d = cc & 63;
          int b = gm >> 11, n = gm & (Nk - 1);
          bf16* dst = (part == 0) ? qo : (part == 1) ? ko : vo;
          dst[(size_t)((b * Hk + h) * Nk + n) * Dk + d] = (bf16)val;
        } else if (MODE == 1) {
          int h = gc >> 6, d = gc & 63;
          int b = gm >> 11, n = gm & (Nk - 1);
          fo[(size_t)((b * Hk + h) * Nk + n) * Dk + d] = val;
        } else {
          fo[(size_t)gm * Ck + gc] = val;
        }
      }
    }
  }
}

// ---------------- flash attention, one wave per 16-query block --------------
// STAGE 1: softmax(QK^T * D^-0.5) V
// STAGE 2: softmax(mask(10 * T T^T)) V  with  s<0 -> -inf
// Output scattered as bf16 into gathered [M=b*N+n, C=h*64+d] for the next GEMM.
template<int STAGE>
__global__ __launch_bounds__(32)
void attn_kernel(const bf16* __restrict__ Tq, const bf16* __restrict__ Tk,
                 const bf16* __restrict__ vT, bf16* __restrict__ outg) {
  __shared__ bf16 pls[16 * 32];                  // P tile bounce buffer (1KB)
  const int lane    = threadIdx.x & 31;
  const int laneRow = lane & 15;
  const int hi      = lane >> 4;
  const int bh = blockIdx.y;
  const int q0 = blockIdx.x * 16;

  const bf16* qbase = Tq + (size_t)bh * Nk * Dk;
  const bf16* kbase = Tk + (size_t)bh * Nk * Dk;
  const bf16* vtb   = vT + (size_t)bh * Dk * Nk;

  const bf16* qrow = qbase + (size_t)(q0 + laneRow) * Dk + hi * 8;
  v16bf qa0 = load_a_frag(qrow);                 // K-dim 0..31 of D=64
  v16bf qa1 = load_a_frag(qrow + 32);            // K-dim 32..63

  v8f acc0 = {}, acc1 = {}, acc2 = {}, acc3 = {};
  float mR[8], lR[8];
#pragma unroll
  for (int r = 0; r < 8; ++r) { mR[r] = -1e30f; lR[r] = 0.f; }

  for (int j0 = 0; j0 < Nk; j0 += 32) {
    // S = Q @ K^T for 32 keys: two 16x16 tiles, K-dim=64 via chained wmma
    const bf16* kr0 = kbase + (size_t)(j0 + laneRow) * Dk + hi * 8;
    const bf16* kr1 = kbase + (size_t)(j0 + 16 + laneRow) * Dk + hi * 8;
    v8f s0 = {}, s1 = {};
    s0 = wmma_bf16(qa0, load_b_frag(kr0), s0);
    s0 = wmma_bf16(qa1, load_b_frag(kr0 + 32), s0);
    s1 = wmma_bf16(qa0, load_b_frag(kr1), s1);
    s1 = wmma_bf16(qa1, load_b_frag(kr1 + 32), s1);

#pragma unroll
    for (int r = 0; r < 8; ++r) {
      float a = s0[r], b = s1[r];
      if (STAGE == 1) { a *= 0.125f; b *= 0.125f; }          // D^-0.5
      else {
        a *= 10.f; b *= 10.f;                                // SCRA scaling
        a = (a < 0.f) ? -1e30f : a;                          // mask negatives
        b = (b < 0.f) ? -1e30f : b;
      }
      // row max across the 16 lanes holding this row's 32 columns
      float rm = fmaxf(a, b);
      rm = fmaxf(rm, __shfl_xor(rm, 1));
      rm = fmaxf(rm, __shfl_xor(rm, 2));
      rm = fmaxf(rm, __shfl_xor(rm, 4));
      rm = fmaxf(rm, __shfl_xor(rm, 8));
      float mN = fmaxf(mR[r], rm);
      float alpha = __expf(mR[r] - mN);
      mR[r] = mN;
      float p0 = __expf(a - mN);
      float p1 = __expf(b - mN);
      if (STAGE == 2) {                                      // hard-zero masked
        p0 = (a <= -1e29f) ? 0.f : p0;
        p1 = (b <= -1e29f) ? 0.f : p1;
      }
      float rs = p0 + p1;
      rs += __shfl_xor(rs, 1);  rs += __shfl_xor(rs, 2);
      rs += __shfl_xor(rs, 4);  rs += __shfl_xor(rs, 8);
      lR[r] = lR[r] * alpha + rs;
      acc0[r] *= alpha; acc1[r] *= alpha; acc2[r] *= alpha; acc3[r] *= alpha;
      // D-layout -> LDS (row-major 16x32) so we can re-read as an A-fragment
      int prow = r + 8 * hi;
      pls[prow * 32 + laneRow]      = (bf16)p0;
      pls[prow * 32 + 16 + laneRow] = (bf16)p1;
    }
    __syncthreads();
    v16bf pa = load_a_frag(&pls[laneRow * 32 + hi * 8]);     // P as A (16x32)
    // V block as B: vT row = d (lane col), keys j0 + hi*8 + e (e<8), j0 + 16 + hi*8 + (e-8) (e>=8)
    const bf16* vr = vtb + (size_t)laneRow * Nk + j0 + hi * 8;
    acc0 = wmma_bf16(pa, load_b_frag(vr),                   acc0);
    acc1 = wmma_bf16(pa, load_b_frag(vr + 16 * (size_t)Nk), acc1);
    acc2 = wmma_bf16(pa, load_b_frag(vr + 32 * (size_t)Nk), acc2);
    acc3 = wmma_bf16(pa, load_b_frag(vr + 48 * (size_t)Nk), acc3);
    __syncthreads();
  }

  const int b = bh / Hk, h = bh % Hk;
#pragma unroll
  for (int r = 0; r < 8; ++r) {
    int n = q0 + r + 8 * hi;
    float inv = 1.0f / lR[r];
    size_t base = (size_t)(b * Nk + n) * Ck + h * Dk;        // gathered [M,C]
    outg[base + 0  + laneRow] = (bf16)(acc0[r] * inv);
    outg[base + 16 + laneRow] = (bf16)(acc1[r] * inv);
    outg[base + 32 + laneRow] = (bf16)(acc2[r] * inv);
    outg[base + 48 + laneRow] = (bf16)(acc3[r] * inv);
  }
}

// ---------------------------------------------------------------------------
extern "C" void kernel_launch(void* const* d_in, const int* in_sizes, int n_in,
                              void* d_out, int out_size, void* d_ws,
                              size_t ws_size, hipStream_t stream) {
  (void)in_sizes; (void)n_in; (void)out_size; (void)ws_size;
  const float* x      = (const float*)d_in[0];   // [B,N,C] (== [M,C], m=b*N+n)
  const float* qkv_w  = (const float*)d_in[1];   // [3C,C]
  const float* qkv_b  = (const float*)d_in[2];   // [3C]
  const float* proj_w = (const float*)d_in[3];   // [C,C]
  const float* proj_b = (const float*)d_in[4];   // [C]
  float* outp = (float*)d_out;                   // [B,N,C]

  char* ws = (char*)d_ws;
  auto take = [&](size_t bytes) {
    char* p = ws;
    ws += (bytes + 255) & ~(size_t)255;
    return p;
  };
  bf16*  xb  = (bf16*)take((size_t)Mk * Ck * 2);
  bf16*  qwb = (bf16*)take((size_t)C3 * Ck * 2);
  bf16*  pwb = (bf16*)take((size_t)Ck * Ck * 2);
  bf16*  qb  = (bf16*)take((size_t)BH * Nk * Dk * 2);
  bf16*  kb  = (bf16*)take((size_t)BH * Nk * Dk * 2);
  bf16*  vb  = (bf16*)take((size_t)BH * Nk * Dk * 2);
  bf16*  vTb = (bf16*)take((size_t)BH * Dk * Nk * 2);
  bf16*  a1  = (bf16*)take((size_t)Mk * Ck * 2);
  float* p1  = (float*)take((size_t)BH * Nk * Dk * 4);
  bf16*  tb  = (bf16*)take((size_t)BH * Nk * Dk * 2);
  bf16*  a2  = (bf16*)take((size_t)Mk * Ck * 2);

  // 1) convert operands to bf16
  f2bf_kernel<<<(Mk * Ck + 255) / 256, 256, 0, stream>>>(x, xb, Mk * Ck);
  f2bf_kernel<<<(C3 * Ck + 255) / 256, 256, 0, stream>>>(qkv_w, qwb, C3 * Ck);
  f2bf_kernel<<<(Ck * Ck + 255) / 256, 256, 0, stream>>>(proj_w, pwb, Ck * Ck);

  // 2) QKV projection -> q/k/v [BH,N,D] bf16
  gemm_kernel<0><<<dim3(C3 / 128, Mk / 64), 256, 0, stream>>>(
      xb, qwb, qkv_b, Ck, qb, kb, vb, nullptr);

  // 3) V^T for the P@V B-operand
  vtrans_kernel<<<(BH * Nk * Dk + 255) / 256, 256, 0, stream>>>(vb, vTb);

  // 4) stage-1 attention -> gathered [M,C] bf16
  attn_kernel<1><<<dim3(Nk / 16, BH), 32, 0, stream>>>(qb, kb, vTb, a1);

  // 5) first projection -> p1 [BH,N,D] f32
  gemm_kernel<1><<<dim3(Ck / 128, Mk / 64), 256, 0, stream>>>(
      a1, pwb, proj_b, Ck, nullptr, nullptr, nullptr, p1);

  // 6) L2-normalize tokens -> tb bf16
  normalize_kernel<<<BH * Nk / 8, 256, 0, stream>>>(p1, tb);

  // 7) stage-2 self-correlation attention (masked) -> gathered [M,C]
  attn_kernel<2><<<dim3(Nk / 16, BH), 32, 0, stream>>>(tb, tb, vTb, a2);

  // 8) second projection -> final output [B,N,C] f32
  gemm_kernel<2><<<dim3(Ck / 128, Mk / 64), 256, 0, stream>>>(
      a2, pwb, proj_b, Ck, nullptr, nullptr, nullptr, outp);
}
